// LagLlamaModel_77670188581138
// MI455X (gfx1250) — matrix operand order repaired and checked by reading, not compiled, not run on hardware
//
#include <hip/hip_runtime.h>
#include <hip/hip_bf16.h>

// ---------------------------------------------------------------------------
// LagLlama forward for MI455X (gfx1250, wave32, WMMA).
// All GEMMs via v_wmma_f32_16x16x32_bf16 (bf16 A/B, fp32 accum).
// GEMM: 128x128 block tile, 8 waves x (32x64) wave tiles, software-pipelined
// global->reg->LDS staging so tile loads issue in one burst per K-step.
// ---------------------------------------------------------------------------

typedef __attribute__((ext_vector_type(16))) __bf16 v16bf;
typedef __attribute__((ext_vector_type(8)))  float  v8f;

static constexpr int B_   = 2;
static constexpr int T_   = 1024;
static constexpr int C_   = 1024;
static constexpr int NH_  = 16;
static constexpr int DH_  = 64;
static constexpr int L_   = 4;
static constexpr int HFF_ = 2816;
static constexpr int FEAT_= 9;
static constexpr int M_   = B_ * T_;       // 2048 GEMM rows

__device__ __forceinline__ v8f vzero8() { v8f z = {}; return z; }

__device__ __forceinline__ v8f wmma_bf16(v16bf a, v16bf b, v8f c) {
    // (neg_a, A, neg_b, B, c_mod, C, reuse_a, reuse_b)
    return __builtin_amdgcn_wmma_f32_16x16x32_bf16(false, a, false, b,
                                                   (short)0, c, false, false);
}

// reductions across each 16-lane half of a wave32 (xor masks stay within half)
__device__ __forceinline__ float half16_max(float v) {
    #pragma unroll
    for (int m = 1; m < 16; m <<= 1) v = fmaxf(v, __shfl_xor(v, m, 32));
    return v;
}
__device__ __forceinline__ float half16_sum(float v) {
    #pragma unroll
    for (int m = 1; m < 16; m <<= 1) v += __shfl_xor(v, m, 32);
    return v;
}

// ---------------------------------------------------------------------------
// 1) input projection: h[m][c] = sum_f x[m][f] * w_in[c][f]   (FEAT=9, tiny)
// ---------------------------------------------------------------------------
__global__ void k_inproj(const float* __restrict__ x,
                         const float* __restrict__ w_in,
                         float* __restrict__ h) {
    int idx = blockIdx.x * 256 + threadIdx.x;        // < M_*C_
    int m = idx >> 10;                               // C_ == 1024
    int c = idx & (C_ - 1);
    const float* xr = x + (size_t)m * FEAT_;
    const float* wr = w_in + (size_t)c * FEAT_;
    float s = 0.f;
    #pragma unroll
    for (int f = 0; f < FEAT_; ++f) s += xr[f] * wr[f];
    h[idx] = s;
}

// ---------------------------------------------------------------------------
// 2) RMSNorm: one 256-thread block per row of length Ccols
// ---------------------------------------------------------------------------
__global__ void k_rmsnorm(const float* __restrict__ x,
                          const float* __restrict__ g,
                          float* __restrict__ out, int Ccols) {
    __shared__ float red[256];
    const int row = blockIdx.x;
    const int tid = threadIdx.x;
    const float* xr = x + (size_t)row * Ccols;
    float s = 0.f;
    for (int i = tid; i < Ccols; i += 256) { float v = xr[i]; s += v * v; }
    red[tid] = s;
    __syncthreads();
    #pragma unroll
    for (int off = 128; off > 0; off >>= 1) {
        if (tid < off) red[tid] += red[tid + off];
        __syncthreads();
    }
    const float inv = rsqrtf(red[0] / (float)Ccols + 1e-5f);
    float* orow = out + (size_t)row * Ccols;
    for (int i = tid; i < Ccols; i += 256) orow[i] = xr[i] * inv * g[i];
}

// ---------------------------------------------------------------------------
// 3) bf16 WMMA GEMM:  D[M,N] (+)= A[M,K] @ W[N,K]^T
//    block tile 128x128, 8 waves of 32x64 (2x4 wmma 16x16x32), K-step 32.
//    Pipelined: next K-tile global loads issue before current-tile compute.
// ---------------------------------------------------------------------------
__global__ void __launch_bounds__(256)
k_gemm(const float* __restrict__ A, const float* __restrict__ W,
       float* __restrict__ D, int N, int K, int addFlag) {
    __shared__ __bf16 As[128][40];   // padded stride -> ds_load_b128-friendly
    __shared__ __bf16 Bs[128][40];

    const int tid  = threadIdx.x;
    const int lane = tid & 31;
    const int wave = tid >> 5;
    const int wtm  = wave >> 1;      // 0..3 : wave row tile (32 rows)
    const int wtn  = wave & 1;       // 0..1 : wave col tile (64 cols)
    const int lr   = lane & 15;      // row/col within frag
    const int kg   = lane >> 4;      // half-wave group
    const int rowBase = blockIdx.y * 128;
    const int colBase = blockIdx.x * 128;

    // staging slots: thread covers rows (tid>>3)+i*32, cols (tid&7)*4 .. +3
    const int sRow = tid >> 3;
    const int sCol = (tid & 7) * 4;

    v8f acc[2][4];
    #pragma unroll
    for (int i = 0; i < 2; ++i)
        #pragma unroll
        for (int j = 0; j < 4; ++j) acc[i][j] = vzero8();

    float4 ra[4], rb[4];
    // prologue: load first K-tile into registers
    #pragma unroll
    for (int i = 0; i < 4; ++i) {
        ra[i] = *(const float4*)(A + (size_t)(rowBase + sRow + i * 32) * K + sCol);
        rb[i] = *(const float4*)(W + (size_t)(colBase + sRow + i * 32) * K + sCol);
    }

    for (int k0 = 0; k0 < K; k0 += 32) {
        // convert + store staged registers to LDS
        #pragma unroll
        for (int i = 0; i < 4; ++i) {
            const int r = sRow + i * 32;
            As[r][sCol + 0] = (__bf16)ra[i].x; As[r][sCol + 1] = (__bf16)ra[i].y;
            As[r][sCol + 2] = (__bf16)ra[i].z; As[r][sCol + 3] = (__bf16)ra[i].w;
            Bs[r][sCol + 0] = (__bf16)rb[i].x; Bs[r][sCol + 1] = (__bf16)rb[i].y;
            Bs[r][sCol + 2] = (__bf16)rb[i].z; Bs[r][sCol + 3] = (__bf16)rb[i].w;
        }
        __syncthreads();

        // issue next K-tile loads (latency hides behind WMMA compute below)
        if (k0 + 32 < K) {
            const int kn = k0 + 32;
            #pragma unroll
            for (int i = 0; i < 4; ++i) {
                ra[i] = *(const float4*)(A + (size_t)(rowBase + sRow + i * 32) * K + kn + sCol);
                rb[i] = *(const float4*)(W + (size_t)(colBase + sRow + i * 32) * K + kn + sCol);
            }
            if (kn + 32 < K) {
                __builtin_prefetch(A + (size_t)(rowBase + sRow) * K + kn + 32 + sCol, 0, 3);
                __builtin_prefetch(W + (size_t)(colBase + sRow) * K + kn + 32 + sCol, 0, 3);
            }
        }

        // build fragments in contiguous k order (k = e + 16*kg) for A and B
        v16bf af[2], bfr[4];
        #pragma unroll
        for (int mi = 0; mi < 2; ++mi) {
            const int r = wtm * 32 + mi * 16 + lr;
            #pragma unroll
            for (int e = 0; e < 16; ++e) {
                const int kk = e + kg * 16;                     // A 16x32 layout
                af[mi][e] = As[r][kk];
            }
        }
        #pragma unroll
        for (int ni = 0; ni < 4; ++ni) {
            const int cc = wtn * 64 + ni * 16 + lr;
            #pragma unroll
            for (int e = 0; e < 16; ++e) {
                const int kk = e + kg * 16;                     // B 32x16 layout
                bfr[ni][e] = Bs[cc][kk];
            }
        }
        #pragma unroll
        for (int mi = 0; mi < 2; ++mi)
            #pragma unroll
            for (int ni = 0; ni < 4; ++ni)
                acc[mi][ni] = wmma_bf16(af[mi], bfr[ni], acc[mi][ni]);
        __syncthreads();
    }

    // epilogue: C/D layout row = r + kg*8, col = lr
    #pragma unroll
    for (int mi = 0; mi < 2; ++mi)
        #pragma unroll
        for (int ni = 0; ni < 4; ++ni)
            #pragma unroll
            for (int r = 0; r < 8; ++r) {
                const int row = rowBase + wtm * 32 + mi * 16 + r + kg * 8;
                const int col = colBase + wtn * 64 + ni * 16 + lr;
                const size_t off = (size_t)row * N + col;
                float val = acc[mi][ni][r];
                if (addFlag) val += D[off];
                D[off] = val;
            }
}

// ---------------------------------------------------------------------------
// 4) RoPE + split/transpose: qbuf[B,T,C], kvbuf[B,T,2C] -> q,k,v [B*NH,T,DH]
// ---------------------------------------------------------------------------
__global__ void k_rope(const float* __restrict__ qb, const float* __restrict__ kvb,
                       float* __restrict__ q, float* __restrict__ k,
                       float* __restrict__ v) {
    const int idx = blockIdx.x * 256 + threadIdx.x;  // ((b*NH+h)*T+t)*DH+d
    const int d  = idx & (DH_ - 1);
    const int t  = (idx >> 6) & (T_ - 1);
    const int hh = (idx >> 16) & (NH_ - 1);
    const int b  = idx >> 20;
    const size_t mrow = (size_t)b * T_ + t;
    const float* qrow = qb  + mrow * C_       + hh * DH_;
    const float* krow = kvb + mrow * (2 * C_) + hh * DH_;
    const float* vrow = kvb + mrow * (2 * C_) + C_ + hh * DH_;
    const int j = d & 31;
    const float inv_freq = __powf(10000.0f, -(float)j * (1.0f / 32.0f));
    const float ang = (float)t * inv_freq;
    const float cs = __cosf(ang), sn = __sinf(ang);
    const float qv = qrow[d];
    const float qr = (d < 32) ? -qrow[d + 32] : qrow[d - 32];
    const float kv = krow[d];
    const float kr = (d < 32) ? -krow[d + 32] : krow[d - 32];
    q[idx] = qv * cs + qr * sn;
    k[idx] = kv * cs + kr * sn;
    v[idx] = vrow[d];
}

// ---------------------------------------------------------------------------
// 5) flash attention (causal, online softmax). One wave = 16 query rows,
//    32 keys per step: S = Q@K^T via 4 WMMAs, P@V via 4 WMMAs.
// ---------------------------------------------------------------------------
__global__ void __launch_bounds__(256)
k_flash(const float* __restrict__ Q, const float* __restrict__ Kt,
        const float* __restrict__ V, float* __restrict__ O) {
    __shared__ __bf16 Pl[8][16][40];  // per-wave P-tile scratch (C-layout -> A-layout)

    const int tid  = threadIdx.x;
    const int lane = tid & 31;
    const int wave = tid >> 5;
    const int lr   = lane & 15;
    const int kg   = lane >> 4;
    const int widx = blockIdx.x * 8 + wave;          // 0..2047
    const int qblk = widx & (T_ / 16 - 1);           // 64 q-blocks
    const int bh   = widx >> 6;                      // 0..31
    const int qBase = qblk * 16;

    const float* Qp = Q  + (size_t)bh * T_ * DH_;
    const float* Kp = Kt + (size_t)bh * T_ * DH_;
    const float* Vp = V  + (size_t)bh * T_ * DH_;

    // Q fragments (16x64 split into two 16x32 A-frags)
    v16bf aq0, aq1;
    {
        const float* qrow = Qp + (size_t)(qBase + lr) * DH_;
        #pragma unroll
        for (int e = 0; e < 16; ++e) {
            const int kk = e + kg * 16;
            aq0[e] = (__bf16)qrow[kk];
            aq1[e] = (__bf16)qrow[32 + kk];
        }
    }

    float mrow[8], lrow[8];
    v8f o[4];
    #pragma unroll
    for (int r = 0; r < 8; ++r) { mrow[r] = -1e30f; lrow[r] = 0.f; }
    #pragma unroll
    for (int n = 0; n < 4; ++n) o[n] = vzero8();

    const float scale = 0.125f;                      // 1/sqrt(64)
    const int smax = ((qBase + 16 + 31) / 32) * 32;  // causal horizon

    for (int s0 = 0; s0 < smax; s0 += 32) {
        float sv[2][8];
        #pragma unroll
        for (int ti = 0; ti < 2; ++ti) {
            v16bf bk0, bk1;
            const float* krow = Kp + (size_t)(s0 + ti * 16 + lr) * DH_;
            #pragma unroll
            for (int e = 0; e < 16; ++e) {
                const int dd = e + kg * 16;          // B 32x16 layout over DH
                bk0[e] = (__bf16)krow[dd];
                bk1[e] = (__bf16)krow[32 + dd];
            }
            v8f s = vzero8();
            s = wmma_bf16(aq0, bk0, s);
            s = wmma_bf16(aq1, bk1, s);
            #pragma unroll
            for (int r = 0; r < 8; ++r) {
                const int scol = s0 + ti * 16 + lr;
                const int qr_  = qBase + r + kg * 8;
                const float vv = s[r] * scale;
                sv[ti][r] = (scol > qr_) ? -1e30f : vv;
            }
        }
        float alpha[8];
        #pragma unroll
        for (int r = 0; r < 8; ++r) {
            float tmax = half16_max(fmaxf(sv[0][r], sv[1][r]));
            const float mn = fmaxf(mrow[r], tmax);
            alpha[r] = __expf(mrow[r] - mn);
            const float p0 = __expf(sv[0][r] - mn);
            const float p1 = __expf(sv[1][r] - mn);
            lrow[r] = lrow[r] * alpha[r] + half16_sum(p0 + p1);
            mrow[r] = mn;
            Pl[wave][r + kg * 8][lr]      = (__bf16)p0;   // C-layout write
            Pl[wave][r + kg * 8][16 + lr] = (__bf16)p1;
        }
        #pragma unroll
        for (int n = 0; n < 4; ++n)
            #pragma unroll
            for (int r = 0; r < 8; ++r) o[n][r] *= alpha[r];

        // reload P in A-matrix layout (16x32 over the 32 keys of this step)
        v16bf pf;
        #pragma unroll
        for (int e = 0; e < 16; ++e) {
            const int kk = e + kg * 16;
            pf[e] = Pl[wave][lr][kk];
        }
        #pragma unroll
        for (int n = 0; n < 4; ++n) {
            v16bf bv;
            #pragma unroll
            for (int e = 0; e < 16; ++e)
                bv[e] = (__bf16)Vp[(size_t)(s0 + e + kg * 16) * DH_ + n * 16 + lr];
            o[n] = wmma_bf16(pf, bv, o[n]);
        }
    }

    // write merged-head output [B,T,C]
    const int b  = bh >> 4;
    const int hh = bh & (NH_ - 1);
    #pragma unroll
    for (int n = 0; n < 4; ++n)
        #pragma unroll
        for (int r = 0; r < 8; ++r) {
            const int trow = qBase + r + kg * 8;
            const size_t off = ((size_t)b * T_ + trow) * C_ + hh * DH_ + n * 16 + lr;
            O[off] = o[n][r] / lrow[r];
        }
}

// ---------------------------------------------------------------------------
// 6) SiLU gate: u = silu(u) * vg  (in place on u)
// ---------------------------------------------------------------------------
__global__ void k_silugate(float* __restrict__ u, const float* __restrict__ vg) {
    const size_t i = (size_t)blockIdx.x * 256 + threadIdx.x;
    const float x = u[i];
    const float sg = 1.0f / (1.0f + __expf(-x));
    u[i] = x * sg * vg[i];
}

// ---------------------------------------------------------------------------
// driver
// ---------------------------------------------------------------------------
extern "C" void kernel_launch(void* const* d_in, const int* in_sizes, int n_in,
                              void* d_out, int out_size, void* d_ws, size_t ws_size,
                              hipStream_t stream) {
    (void)in_sizes; (void)n_in; (void)out_size; (void)ws_size;
    const float* x    = (const float*)d_in[0];
    const float* w_in = (const float*)d_in[1];
    const float* wq   = (const float*)d_in[2];
    const float* wkv  = (const float*)d_in[3];
    const float* wo   = (const float*)d_in[4];
    const float* w1   = (const float*)d_in[5];
    const float* w2   = (const float*)d_in[6];
    const float* w3   = (const float*)d_in[7];
    const float* g1   = (const float*)d_in[8];
    const float* g2   = (const float*)d_in[9];
    const float* gf   = (const float*)d_in[10];

    const size_t MC = (size_t)M_ * C_;               // 2,097,152 floats
    float* ws   = (float*)d_ws;
    float* h    = ws;                                // [M,C]
    float* ain  = ws + MC;                           // [M,C]
    float* regA = ws + 2 * MC;                       // qbuf[M,C] + kvbuf[M,2C] | u[M,HFF]
    float* regB = regA + 3 * MC;                     // q,k,v [B*NH,T,DH] x3    | vg[M,HFF]
    float* attn = regB + 3 * MC;                     // [M,C]
    float* qbuf = regA;
    float* kvbuf= regA + MC;
    float* ubuf = regA;
    float* vgbuf= regB;
    float* qr   = regB;
    float* kr   = regB + MC;
    float* vr   = regB + 2 * MC;

    const dim3 blk(256);
    const int mTiles = M_ / 128;                     // 16

    // input projection
    k_inproj<<<(M_ * C_) / 256, blk, 0, stream>>>(x, w_in, h);

    for (int l = 0; l < L_; ++l) {
        const float* wql  = wq  + (size_t)l * C_ * C_;
        const float* wkvl = wkv + (size_t)l * 2 * C_ * C_;
        const float* wol  = wo  + (size_t)l * C_ * C_;
        const float* w1l  = w1  + (size_t)l * HFF_ * C_;
        const float* w2l  = w2  + (size_t)l * HFF_ * C_;
        const float* w3l  = w3  + (size_t)l * C_ * HFF_;

        // ---- attention block ----
        k_rmsnorm<<<M_, blk, 0, stream>>>(h, g1 + (size_t)l * C_, ain, C_);
        k_gemm<<<dim3(C_ / 128, mTiles), blk, 0, stream>>>(ain, wql,  qbuf,  C_,     C_, 0);
        k_gemm<<<dim3(2 * C_ / 128, mTiles), blk, 0, stream>>>(ain, wkvl, kvbuf, 2 * C_, C_, 0);
        k_rope<<<(B_ * NH_ * T_ * DH_) / 256, blk, 0, stream>>>(qbuf, kvbuf, qr, kr, vr);
        k_flash<<<(B_ * NH_ * T_ / 16) / 8, blk, 0, stream>>>(qr, kr, vr, attn);
        k_gemm<<<dim3(C_ / 128, mTiles), blk, 0, stream>>>(attn, wol, h, C_, C_, 1);

        // ---- MLP block ----
        k_rmsnorm<<<M_, blk, 0, stream>>>(h, g2 + (size_t)l * C_, ain, C_);
        k_gemm<<<dim3(HFF_ / 128, mTiles), blk, 0, stream>>>(ain, w1l, ubuf,  HFF_, C_, 0);
        k_gemm<<<dim3(HFF_ / 128, mTiles), blk, 0, stream>>>(ain, w2l, vgbuf, HFF_, C_, 0);
        k_silugate<<<((size_t)M_ * HFF_) / 256, blk, 0, stream>>>(ubuf, vgbuf);
        k_gemm<<<dim3(C_ / 128, mTiles), blk, 0, stream>>>(ubuf, w3l, h, C_, HFF_, 1);
    }

    // final norm -> d_out
    k_rmsnorm<<<M_, blk, 0, stream>>>(h, gf, (float*)d_out, C_);
}
